// TGAM_LinkPrediction_Improved_53652731462279
// MI455X (gfx1250) — matrix operand order repaired and checked by reading, not compiled, not run on hardware
//
#include <hip/hip_runtime.h>
#include <math.h>

// ---------------------------------------------------------------------------
// TGAM link prediction on MI455X (gfx1250): all GEMMs + attention run on
// v_wmma_f32_16x16x32_f16 (wave32). fp32 activations between stages, f16
// staged operands, f32 accumulation.
// ---------------------------------------------------------------------------

typedef __attribute__((ext_vector_type(16))) _Float16 v16h;
typedef __attribute__((ext_vector_type(8)))  float    v8f;
typedef __attribute__((ext_vector_type(4)))  unsigned int u32x4;

union Frag { v16h h; u32x4 q[2]; };

#define CEILDIV(a, b) (((a) + (b) - 1) / (b))

#define K_H      256
#define K_NF     128
#define K_EF     128
#define K_MAXN   20000
#define K_E      2048
#define K_NHEAD  8
#define K_DH     32

// ---------------------------------------------------------------------------
// Generic GEMM: C[M,N] = A[M,Kp] * W[N,Kp]^T + bias.  A, W are f16 row-major
// with identical (padded) leading dim Kp (multiple of 32). C is f32 row-major.
// Block: 256 threads = 8 waves, tile M64 x N128, each wave owns 32x32.
// Staging is branch-free (clamped-address b128 loads + cndmask select) and
// software-pipelined: tile k+1 global loads overlap tile k WMMAs.
// ---------------------------------------------------------------------------
__global__ __launch_bounds__(256) void gemm_wmma(
    const _Float16* __restrict__ A, const _Float16* __restrict__ W,
    const float* __restrict__ bias, float* __restrict__ C,
    int M, int N, int Kp) {
  __shared__ unsigned int sA[64 * 16];   // 64 rows x 32 halves (16 dwords)
  __shared__ unsigned int sB[128 * 16];  // 128 rows x 32 halves

  const int tid  = threadIdx.x;
  const int lane = tid & 31;
  const int wave = tid >> 5;
  const int wm   = wave >> 2;   // 0..1 -> M offset wm*32
  const int wn   = wave & 3;    // 0..3 -> N offset wn*32
  const int m0   = blockIdx.y * 64;
  const int n0   = blockIdx.x * 128;
  const int la   = lane & 15;
  const int kb2  = (lane >> 4) << 2;  // A-frag dword sel (interleaved layout)
  const int kb8  = (lane >> 4) << 3;  // unused; B frags also use kb2 (same k order as A)

  const unsigned int* Ad = (const unsigned int*)A;
  const unsigned int* Wd = (const unsigned int*)W;
  const int ldd = Kp >> 1;  // leading dim in dwords (multiple of 16)

  // Staging coordinates: 1 quad (u32x4) of A, 2 quads of B per thread.
  const int qr = tid >> 2;            // row 0..63
  const int qc = (tid & 3) << 2;      // dword col 0,4,8,12
  const int am  = m0 + qr;
  const int bn0 = n0 + qr;
  const int bn1 = n0 + qr + 64;
  const bool aok  = am  < M;
  const bool b0ok = bn0 < N;
  const bool b1ok = bn1 < N;
  const size_t abase  = (size_t)(aok  ? am  : (M - 1)) * ldd + qc;
  const size_t bbase0 = (size_t)(b0ok ? bn0 : (N - 1)) * ldd + qc;
  const size_t bbase1 = (size_t)(b1ok ? bn1 : (N - 1)) * ldd + qc;
  const u32x4 zero4 = {0u, 0u, 0u, 0u};

  v8f acc[2][2];
#pragma unroll
  for (int i = 0; i < 2; ++i)
#pragma unroll
    for (int j = 0; j < 2; ++j)
#pragma unroll
      for (int r = 0; r < 8; ++r) acc[i][j][r] = 0.0f;

  // Prologue fetch (kt = 0): unconditional b128 loads, guarded by select.
  u32x4 ra, rb0, rb1;
  {
    u32x4 t0 = *(const u32x4*)(Ad + abase);
    u32x4 t1 = *(const u32x4*)(Wd + bbase0);
    u32x4 t2 = *(const u32x4*)(Wd + bbase1);
    ra  = aok  ? t0 : zero4;
    rb0 = b0ok ? t1 : zero4;
    rb1 = b1ok ? t2 : zero4;
  }

  for (int kt = 0; kt < Kp; kt += 32) {
    __syncthreads();  // previous tile fully consumed
    *(u32x4*)&sA[qr * 16 + qc] = ra;
    *(u32x4*)&sB[qr * 16 + qc] = rb0;
    *(u32x4*)&sB[(qr + 64) * 16 + qc] = rb1;
    __syncthreads();

    // Prefetch next tile into registers; overlaps the WMMAs below.
    if (kt + 32 < Kp) {
      const size_t k16 = (size_t)((kt + 32) >> 1);
      u32x4 t0 = *(const u32x4*)(Ad + abase + k16);
      u32x4 t1 = *(const u32x4*)(Wd + bbase0 + k16);
      u32x4 t2 = *(const u32x4*)(Wd + bbase1 + k16);
      ra  = aok  ? t0 : zero4;
      rb0 = b0ok ? t1 : zero4;
      rb1 = b1ok ? t2 : zero4;
    }

    Frag a[2], b[2];
#pragma unroll
    for (int ms = 0; ms < 2; ++ms) {
      int mr = wm * 32 + ms * 16 + la;
      a[ms].q[0] = *(const u32x4*)&sA[mr * 16 + kb2];
      a[ms].q[1] = *(const u32x4*)&sA[mr * 16 + 8 + kb2];
    }
#pragma unroll
    for (int ns = 0; ns < 2; ++ns) {
      int nr = wn * 32 + ns * 16 + la;
      b[ns].q[0] = *(const u32x4*)&sB[nr * 16 + kb2];
      b[ns].q[1] = *(const u32x4*)&sB[nr * 16 + 8 + kb2];
    }
#pragma unroll
    for (int ms = 0; ms < 2; ++ms)
#pragma unroll
      for (int ns = 0; ns < 2; ++ns)
        acc[ms][ns] = __builtin_amdgcn_wmma_f32_16x16x32_f16(
            false, a[ms].h, false, b[ns].h, (short)0, acc[ms][ns], false, false);
  }

  // Epilogue: bias add + store.  C layout: vgpr r, lanes<16 -> M=r, else M=8+r.
#pragma unroll
  for (int ms = 0; ms < 2; ++ms)
#pragma unroll
    for (int ns = 0; ns < 2; ++ns) {
      int n = n0 + wn * 32 + ns * 16 + la;
      if (n >= N) continue;
      float bv = bias ? bias[n] : 0.0f;
      int mbase = m0 + wm * 32 + ms * 16 + ((lane >> 4) << 3);
#pragma unroll
      for (int r = 0; r < 8; ++r) {
        int m = mbase + r;
        if (m < M) C[(size_t)m * N + n] = acc[ms][ns][r] + bv;
      }
    }
}

// ---------------------------------------------------------------------------
// Flash attention: 1 wave per (16-query tile, head). qkv f16 [E,768]
// (q cols 0..255, k 256..511, v 512..767). dh=32 -> QK^T is one WMMA per
// 16-key subtile (K-dim = dh = 32). Online softmax, P/V staged via LDS.
// ---------------------------------------------------------------------------
__global__ __launch_bounds__(32) void flash_attn(
    const _Float16* __restrict__ qkv, float* __restrict__ out, int E) {
  __shared__ _Float16 sP[16 * 32];    // P tile: [query m][key j]
  __shared__ _Float16 sVt[32 * 32];   // V^T tile: [dh d][key k]
  const int lane = threadIdx.x;
  const int qt = blockIdx.x, h = blockIdx.y;
  const int la = lane & 15;
  const int kb2 = (lane >> 4) << 2;
  const int kb8 = (lane >> 4) << 3;
  const unsigned int* Qd = (const unsigned int*)qkv;  // row = 384 dwords
  unsigned int* sPd = (unsigned int*)sP;
  unsigned int* sVtd = (unsigned int*)sVt;

  Frag qa;
  {
    int m = qt * 16 + la;
    const unsigned int* p = Qd + (size_t)m * 384 + h * 16;
    qa.q[0] = *(const u32x4*)(p + kb2);
    qa.q[1] = *(const u32x4*)(p + 8 + kb2);
  }
  v8f o0, o1;
  float rmax[8], rsum[8];
#pragma unroll
  for (int r = 0; r < 8; ++r) { o0[r] = 0.f; o1[r] = 0.f; rmax[r] = -1e30f; rsum[r] = 0.f; }
  const float sc = 0.17677669529663687f;  // 1/sqrt(32)

  for (int kt = 0; kt < E; kt += 32) {
    Frag kf0, kf1;
    {
      const unsigned int* p = Qd + (size_t)(kt + la) * 384 + 128 + h * 16;
      kf0.q[0] = *(const u32x4*)(p + kb2);
      kf0.q[1] = *(const u32x4*)(p + 8 + kb2);
    }
    {
      const unsigned int* p = Qd + (size_t)(kt + 16 + la) * 384 + 128 + h * 16;
      kf1.q[0] = *(const u32x4*)(p + kb2);
      kf1.q[1] = *(const u32x4*)(p + 8 + kb2);
    }
    // Stage V^T for this key tile (b128 loads, lane = key): issued before the
    // score WMMAs so the global loads hide under matrix work.
    {
      const u32x4* p = (const u32x4*)(Qd + (size_t)(kt + lane) * 384 + 256 + h * 16);
      union { u32x4 q; unsigned int u[4]; } blk;
#pragma unroll
      for (int j = 0; j < 4; ++j) {
        blk.q = p[j];
#pragma unroll
        for (int i = 0; i < 4; ++i) {
          union { unsigned int u; _Float16 f[2]; } cv;
          cv.u = blk.u[i];
          int d = (j * 4 + i) * 2;
          sVt[d * 32 + lane] = cv.f[0];
          sVt[(d + 1) * 32 + lane] = cv.f[1];
        }
      }
    }
    v8f s0, s1;
#pragma unroll
    for (int r = 0; r < 8; ++r) { s0[r] = 0.f; s1[r] = 0.f; }
    s0 = __builtin_amdgcn_wmma_f32_16x16x32_f16(false, qa.h, false, kf0.h, (short)0, s0, false, false);
    s1 = __builtin_amdgcn_wmma_f32_16x16x32_f16(false, qa.h, false, kf1.h, (short)0, s1, false, false);

    // Online softmax per C-fragment row (rows split across lane halves).
#pragma unroll
    for (int r = 0; r < 8; ++r) {
      float a0 = s0[r] * sc, a1 = s1[r] * sc;
      float mx = fmaxf(a0, a1);
#pragma unroll
      for (int off = 1; off < 16; off <<= 1) mx = fmaxf(mx, __shfl_xor(mx, off, 32));
      float nm = fmaxf(rmax[r], mx);
      float al = expf(rmax[r] - nm);
      float p0 = expf(a0 - nm), p1 = expf(a1 - nm);
      float tsum = p0 + p1;
#pragma unroll
      for (int off = 1; off < 16; off <<= 1) tsum += __shfl_xor(tsum, off, 32);
      rsum[r] = rsum[r] * al + tsum;
      rmax[r] = nm;
      o0[r] *= al; o1[r] *= al;
      int m = r + ((lane >> 4) << 3);
      sP[m * 32 + la] = (_Float16)p0;
      sP[m * 32 + 16 + la] = (_Float16)p1;
    }
    __syncthreads();
    Frag pf, v0, v1;
    pf.q[0] = *(const u32x4*)&sPd[la * 16 + kb2];
    pf.q[1] = *(const u32x4*)&sPd[la * 16 + 8 + kb2];
    v0.q[0] = *(const u32x4*)&sVtd[la * 16 + kb2];
    v0.q[1] = *(const u32x4*)&sVtd[la * 16 + 8 + kb2];
    v1.q[0] = *(const u32x4*)&sVtd[(16 + la) * 16 + kb2];
    v1.q[1] = *(const u32x4*)&sVtd[(16 + la) * 16 + 8 + kb2];
    o0 = __builtin_amdgcn_wmma_f32_16x16x32_f16(false, pf.h, false, v0.h, (short)0, o0, false, false);
    o1 = __builtin_amdgcn_wmma_f32_16x16x32_f16(false, pf.h, false, v1.h, (short)0, o1, false, false);
    __syncthreads();
  }
#pragma unroll
  for (int r = 0; r < 8; ++r) {
    float inv = 1.0f / rsum[r];
    int m = qt * 16 + r + ((lane >> 4) << 3);
    out[(size_t)m * 256 + h * 32 + la] = o0[r] * inv;
    out[(size_t)m * 256 + h * 32 + 16 + la] = o1[r] * inv;
  }
}

// ---------------------------------------------------------------------------
// Elementwise / utility kernels
// ---------------------------------------------------------------------------
// f32 -> f16 segment convert with optional zero tail (c in [0,Cz), valid < C).
__global__ void cvt_seg(const float* __restrict__ src, int ld_src,
                        _Float16* __restrict__ dst, int ldd, int coloff,
                        int C, int Cz, int M) {
  int idx = blockIdx.x * blockDim.x + threadIdx.x;
  if (idx >= M * Cz) return;
  int r = idx / Cz, c = idx - r * Cz;
  dst[(size_t)r * ldd + coloff + c] =
      (c < C) ? (_Float16)src[(size_t)r * ld_src + c] : (_Float16)0.0f;
}

__global__ void gather_cvt(const float* __restrict__ table, int tld,
                           const int* __restrict__ iv,
                           _Float16* __restrict__ dst, int ldd, int coloff,
                           int C, int E) {
  int idx = blockIdx.x * blockDim.x + threadIdx.x;
  if (idx >= E * C) return;
  int r = idx / C, c = idx - r * C;
  int row = iv[r];
  dst[(size_t)r * ldd + coloff + c] = (_Float16)table[(size_t)row * tld + c];
}

// LayerNorm over rows (optionally y = LN(a+b)), optional leaky-relu(0.1).
__global__ __launch_bounds__(256) void ln_rows(
    const float* __restrict__ xin, const float* __restrict__ add,
    float* __restrict__ out, const float* __restrict__ g,
    const float* __restrict__ b, int C, int do_lrelu) {
  __shared__ float red[256];
  int row = blockIdx.x, tid = threadIdx.x;
  float vals[4];
  int cnt = 0;
  float s = 0.f;
  for (int c = tid; c < C; c += 256) {
    float v = xin[(size_t)row * C + c];
    if (add) v += add[(size_t)row * C + c];
    vals[cnt++] = v;
    s += v;
  }
  red[tid] = s; __syncthreads();
  for (int o = 128; o > 0; o >>= 1) { if (tid < o) red[tid] += red[tid + o]; __syncthreads(); }
  float mean = red[0] / C;
  __syncthreads();
  float s2 = 0.f; cnt = 0;
  for (int c = tid; c < C; c += 256) { float d = vals[cnt++] - mean; s2 += d * d; }
  red[tid] = s2; __syncthreads();
  for (int o = 128; o > 0; o >>= 1) { if (tid < o) red[tid] += red[tid + o]; __syncthreads(); }
  float inv = rsqrtf(red[0] / C + 1e-5f);
  cnt = 0;
  for (int c = tid; c < C; c += 256) {
    float y = (vals[cnt++] - mean) * inv * g[c] + b[c];
    if (do_lrelu) y = (y > 0.f) ? y : 0.1f * y;
    out[(size_t)row * C + c] = y;
  }
}

// BatchNorm (eval, fresh stats): y = lrelu(x * g / sqrt(1+eps) + b)
__global__ void bn_lrelu(float* __restrict__ x, const float* __restrict__ g,
                         const float* __restrict__ b, int total, int C) {
  int idx = blockIdx.x * blockDim.x + threadIdx.x;
  if (idx >= total) return;
  int c = idx % C;
  float y = x[idx] * rsqrtf(1.0f + 1e-5f) * g[c] + b[c];
  x[idx] = (y > 0.f) ? y : 0.1f * y;
}

__global__ void gelu_k(float* __restrict__ x, int total) {
  int idx = blockIdx.x * blockDim.x + threadIdx.x;
  if (idx >= total) return;
  float v = x[idx];
  x[idx] = 0.5f * v * (1.0f + erff(v * 0.70710678118654752f));
}

__global__ void add_inplace(float* __restrict__ a, const float* __restrict__ b, int total) {
  int idx = blockIdx.x * blockDim.x + threadIdx.x;
  if (idx < total) a[idx] += b[idx];
}

__global__ void scatter_add2(float* __restrict__ dst, const int* __restrict__ di,
                             const float* __restrict__ mf, const int* __restrict__ si,
                             const float* __restrict__ mb, int E, int C) {
  int idx = blockIdx.x * blockDim.x + threadIdx.x;
  if (idx >= E * C) return;
  int e = idx / C, c = idx - e * C;
  atomicAdd(&dst[(size_t)di[e] * C + c], mf[idx]);
  atomicAdd(&dst[(size_t)si[e] * C + c], mb[idx]);
}

__global__ void posenc(float* __restrict__ x, int E, int H) {
  int idx = blockIdx.x * blockDim.x + threadIdx.x;
  if (idx >= E * H) return;
  int e = idx / H, c = idx - e * H;
  float i2 = (float)(c & ~1);
  float freq = expf(-logf(10000.0f) * i2 / (float)H);
  float arg = (float)e * freq;
  x[idx] += (c & 1) ? cosf(arg) : sinf(arg);
}

// temporal input: col 128 = ts, cols 129..159 = 0 (K padded 129 -> 160)
__global__ void te_fill(_Float16* __restrict__ dst, const float* __restrict__ ts, int E) {
  int e = blockIdx.x * blockDim.x + threadIdx.x;
  if (e >= E) return;
  _Float16* row = dst + (size_t)e * 160;
  row[128] = (_Float16)ts[e];
  for (int c = 129; c < 160; ++c) row[c] = (_Float16)0.0f;
}

__global__ void lp_concat(const float* __restrict__ x, const float* __restrict__ nemb,
                          const int* __restrict__ srcl, const int* __restrict__ cand,
                          float* __restrict__ z, int E) {
  int i = threadIdx.x;  // 256 threads
  int s = srcl[E - 1], d = cand[0];
  z[i]       = x[(size_t)(E - 1) * 256 + i];
  z[256 + i] = nemb[(size_t)s * 256 + i];
  z[512 + i] = nemb[(size_t)d * 256 + i];
}

__global__ __launch_bounds__(128) void gemv_k(const float* __restrict__ W,
                                              const float* __restrict__ x,
                                              const float* __restrict__ b,
                                              float* __restrict__ y, int K) {
  __shared__ float red[128];
  int n = blockIdx.x, tid = threadIdx.x;
  float s = 0.f;
  for (int k = tid; k < K; k += 128) s += W[(size_t)n * K + k] * x[k];
  red[tid] = s; __syncthreads();
  for (int o = 64; o > 0; o >>= 1) { if (tid < o) red[tid] += red[tid + o]; __syncthreads(); }
  if (tid == 0) y[n] = red[0] + b[n];
}

__global__ __launch_bounds__(128) void lp_final(const float* __restrict__ W4,
                                                const float* __restrict__ h,
                                                const float* __restrict__ b4,
                                                float* __restrict__ out) {
  __shared__ float red[128];
  int tid = threadIdx.x;
  red[tid] = W4[tid] * h[tid]; __syncthreads();
  for (int o = 64; o > 0; o >>= 1) { if (tid < o) red[tid] += red[tid + o]; __syncthreads(); }
  if (tid == 0) out[0] = 1.0f / (1.0f + expf(-(red[0] + b4[0])));
}

// ---------------------------------------------------------------------------
extern "C" void kernel_launch(void* const* d_in, const int* in_sizes, int n_in,
                              void* d_out, int out_size, void* d_ws, size_t ws_size,
                              hipStream_t stream) {
  (void)in_sizes; (void)n_in; (void)out_size; (void)ws_size;
  const int* src_l = (const int*)d_in[0];
  const int* dst_l = (const int*)d_in[1];
  const int* cand  = (const int*)d_in[2];
  const float* ts  = (const float*)d_in[3];
  const int* eidx  = (const int*)d_in[4];
  const float* nodef = (const float*)d_in[5];
  const float* edgef = (const float*)d_in[6];
  auto P = [&](int i) { return (const float*)d_in[7 + i]; };
  // param base offsets (insertion order of setup_inputs()['params'])
  const int NE = 0, EE = 8, MSG = 16, AGG = 28, TE = 34, TLb = 42, LP = 90;

  // ---- workspace layout (bump allocator, 256B aligned) ----
  char* wp = (char*)d_ws;
  auto alloc = [&](size_t bytes) { char* p = wp; wp += (bytes + 255) & ~(size_t)255; return p; };
  float*    NA    = (float*)alloc((size_t)K_MAXN * K_H * 4);
  float*    NB    = (float*)alloc((size_t)K_MAXN * K_H * 4);   // node_emb
  float*    NC    = (float*)alloc((size_t)K_MAXN * K_H * 4);   // new_emb / agg out
  _Float16* NH    = (_Float16*)alloc((size_t)K_MAXN * K_H * 2);
  _Float16* W16   = (_Float16*)alloc(1 << 20);                 // f16 weight scratch
  _Float16* BIG16 = (_Float16*)alloc((size_t)K_E * 1024 * 2);  // Z / qkv16 / ff16
  _Float16* E16   = (_Float16*)alloc((size_t)K_E * 256 * 2);
  _Float16* TE16  = (_Float16*)alloc((size_t)K_E * 160 * 2);
  float*    EDGE  = (float*)alloc((size_t)K_E * 256 * 4);
  float*    EH    = (float*)alloc((size_t)K_E * 256 * 4);
  float*    MFWD  = (float*)alloc((size_t)K_E * 256 * 4);
  float*    MBWD  = (float*)alloc((size_t)K_E * 256 * 4);
  float*    X     = (float*)alloc((size_t)K_E * 256 * 4);
  float*    QKV   = (float*)alloc((size_t)K_E * 768 * 4);
  float*    ATTN  = (float*)alloc((size_t)K_E * 256 * 4);
  float*    T1    = (float*)alloc((size_t)K_E * 1024 * 4);
  float*    T2    = (float*)alloc((size_t)K_E * 256 * 4);
  float*    LPZ   = (float*)alloc(4096);
  float*    LPH1  = (float*)alloc(2048);
  float*    LPH2  = (float*)alloc(1024);
  float*    LPH3  = (float*)alloc(512);

  auto gemm = [&](const _Float16* A, const _Float16* W, const float* bias,
                  float* C, int M, int N, int Kp) {
    dim3 g(CEILDIV(N, 128), CEILDIV(M, 64));
    gemm_wmma<<<g, 256, 0, stream>>>(A, W, bias, C, M, N, Kp);
  };
  auto cvt = [&](const float* s, int lds_, _Float16* d, int ldd, int co, int C, int Cz, int M) {
    cvt_seg<<<CEILDIV(M * Cz, 256), 256, 0, stream>>>(s, lds_, d, ldd, co, C, Cz, M);
  };
  auto gat = [&](const float* t, int tld, const int* iv, _Float16* d, int ldd, int co, int C) {
    gather_cvt<<<CEILDIV(K_E * C, 256), 256, 0, stream>>>(t, tld, iv, d, ldd, co, C, K_E);
  };
  auto lnk = [&](const float* in, const float* add, float* out, const float* g,
                 const float* b, int rows, int C, int lr) {
    ln_rows<<<rows, 256, 0, stream>>>(in, add, out, g, b, C, lr);
  };
  auto bnk = [&](float* x, const float* g, const float* b, int rows, int C) {
    bn_lrelu<<<CEILDIV(rows * C, 256), 256, 0, stream>>>(x, g, b, rows * C, C);
  };

  // ================= node encoder =================
  cvt(nodef, K_NF, NH, K_NF, 0, K_NF, K_NF, K_MAXN);
  cvt(P(NE + 0), K_NF, W16, K_NF, 0, K_NF, K_NF, K_H);
  gemm(NH, W16, P(NE + 1), NA, K_MAXN, K_H, K_NF);
  bnk(NA, P(NE + 2), P(NE + 3), K_MAXN, K_H);
  cvt(NA, K_H, NH, K_H, 0, K_H, K_H, K_MAXN);
  cvt(P(NE + 4), K_H, W16, K_H, 0, K_H, K_H, K_H);
  gemm(NH, W16, P(NE + 5), NB, K_MAXN, K_H, K_H);
  bnk(NB, P(NE + 6), P(NE + 7), K_MAXN, K_H);  // NB = node_emb

  // ================= edge encoder =================
  gat(edgef, K_EF, eidx, E16, K_EF, 0, K_EF);
  cvt(P(EE + 0), K_EF, W16, K_EF, 0, K_EF, K_EF, K_H);
  gemm(E16, W16, P(EE + 1), EH, K_E, K_H, K_EF);
  bnk(EH, P(EE + 2), P(EE + 3), K_E, K_H);
  cvt(EH, K_H, E16, K_H, 0, K_H, K_H, K_E);
  cvt(P(EE + 4), K_H, W16, K_H, 0, K_H, K_H, K_H);
  gemm(E16, W16, P(EE + 5), EDGE, K_E, K_H, K_H);
  bnk(EDGE, P(EE + 6), P(EE + 7), K_E, K_H);  // EDGE = edge_emb

  // ================= message-passing layers =================
  for (int l = 0; l < 2; ++l) {
    const float* W1 = P(MSG + 6 * l + 0);
    const float* b1 = P(MSG + 6 * l + 1);
    const float* lg = P(MSG + 6 * l + 2);
    const float* lb = P(MSG + 6 * l + 3);
    const float* W2 = P(MSG + 6 * l + 4);
    const float* b2 = P(MSG + 6 * l + 5);
    cvt(EDGE, K_H, BIG16, 768, 256, K_H, K_H, K_E);  // middle segment of Z
    cvt(W1, 768, W16, 768, 0, 768, 768, K_H);
    // forward messages: Z = [hs | e | hd]
    gat(NB, K_H, src_l, BIG16, 768, 0, K_H);
    gat(NB, K_H, dst_l, BIG16, 768, 512, K_H);
    gemm(BIG16, W16, b1, EH, K_E, K_H, 768);
    lnk(EH, nullptr, EH, lg, lb, K_E, K_H, 1);
    // backward messages: Z = [hd | e | hs]
    gat(NB, K_H, dst_l, BIG16, 768, 0, K_H);
    gat(NB, K_H, src_l, BIG16, 768, 512, K_H);
    gemm(BIG16, W16, b1, T2, K_E, K_H, 768);
    lnk(T2, nullptr, T2, lg, lb, K_E, K_H, 1);
    // second linear of the message MLP
    cvt(W2, K_H, W16, K_H, 0, K_H, K_H, K_H);
    cvt(EH, K_H, E16, K_H, 0, K_H, K_H, K_E);
    gemm(E16, W16, b2, MFWD, K_E, K_H, K_H);
    cvt(T2, K_H, E16, K_H, 0, K_H, K_H, K_E);
    gemm(E16, W16, b2, MBWD, K_E, K_H, K_H);
    // new_emb = node_emb scatter-add messages
    hipMemcpyAsync(NC, NB, (size_t)K_MAXN * K_H * 4, hipMemcpyDeviceToDevice, stream);
    scatter_add2<<<CEILDIV(K_E * K_H, 256), 256, 0, stream>>>(NC, dst_l, MFWD, src_l, MBWD, K_E, K_H);
    // agg MLP over all nodes; node_emb += agg
    cvt(NC, K_H, NH, K_H, 0, K_H, K_H, K_MAXN);
    cvt(P(AGG + 0), K_H, W16, K_H, 0, K_H, K_H, K_H);
    gemm(NH, W16, P(AGG + 1), NA, K_MAXN, K_H, K_H);
    lnk(NA, nullptr, NA, P(AGG + 2), P(AGG + 3), K_MAXN, K_H, 1);
    cvt(NA, K_H, NH, K_H, 0, K_H, K_H, K_MAXN);
    cvt(P(AGG + 4), K_H, W16, K_H, 0, K_H, K_H, K_H);
    gemm(NH, W16, P(AGG + 5), NC, K_MAXN, K_H, K_H);
    add_inplace<<<CEILDIV(K_MAXN * K_H, 256), 256, 0, stream>>>(NB, NC, K_MAXN * K_H);
  }

  // ================= temporal encoder =================
  gat(edgef, K_EF, eidx, TE16, 160, 0, K_EF);
  te_fill<<<CEILDIV(K_E, 256), 256, 0, stream>>>(TE16, ts, K_E);
  cvt(P(TE + 0), 129, W16, 160, 0, 129, 160, K_H);   // te_W1 padded 129 -> 160
  gemm(TE16, W16, P(TE + 1), T2, K_E, K_H, 160);
  lnk(T2, nullptr, X, P(TE + 2), P(TE + 3), K_E, K_H, 1);
  cvt(X, K_H, E16, K_H, 0, K_H, K_H, K_E);
  cvt(P(TE + 4), K_H, W16, K_H, 0, K_H, K_H, K_H);
  gemm(E16, W16, P(TE + 5), T2, K_E, K_H, K_H);
  lnk(T2, nullptr, X, P(TE + 6), P(TE + 7), K_E, K_H, 1);
  posenc<<<CEILDIV(K_E * K_H, 256), 256, 0, stream>>>(X, K_E, K_H);

  // ================= transformer layers =================
  for (int l = 0; l < 4; ++l) {
    int tb = TLb + 12 * l;
    cvt(X, K_H, E16, K_H, 0, K_H, K_H, K_E);
    cvt(P(tb + 0), K_H, W16, K_H, 0, K_H, K_H, 768);  // Wqkv [768,256]
    gemm(E16, W16, P(tb + 1), QKV, K_E, 768, K_H);
    cvt(QKV, 768, BIG16, 768, 0, 768, 768, K_E);
    flash_attn<<<dim3(K_E / 16, K_NHEAD), 32, 0, stream>>>(BIG16, ATTN, K_E);
    cvt(ATTN, K_H, E16, K_H, 0, K_H, K_H, K_E);
    cvt(P(tb + 2), K_H, W16, K_H, 0, K_H, K_H, K_H);  // Wo
    gemm(E16, W16, P(tb + 3), T2, K_E, K_H, K_H);
    lnk(X, T2, X, P(tb + 4), P(tb + 5), K_E, K_H, 0); // x = LN(x + attn_proj)
    cvt(X, K_H, E16, K_H, 0, K_H, K_H, K_E);
    cvt(P(tb + 6), K_H, W16, K_H, 0, K_H, K_H, 1024); // Wf1 [1024,256]
    gemm(E16, W16, P(tb + 7), T1, K_E, 1024, K_H);
    gelu_k<<<CEILDIV(K_E * 1024, 256), 256, 0, stream>>>(T1, K_E * 1024);
    cvt(T1, 1024, BIG16, 1024, 0, 1024, 1024, K_E);
    cvt(P(tb + 8), 1024, W16, 1024, 0, 1024, 1024, K_H); // Wf2 [256,1024]
    gemm(BIG16, W16, P(tb + 9), T2, K_E, K_H, 1024);
    lnk(X, T2, X, P(tb + 10), P(tb + 11), K_E, K_H, 0);  // x = LN(x + ffn)
  }

  // ================= link predictor =================
  lp_concat<<<1, 256, 0, stream>>>(X, NB, src_l, cand, LPZ, K_E);
  gemv_k<<<512, 128, 0, stream>>>(P(LP + 0), LPZ, P(LP + 1), LPH1, 768);
  lnk(LPH1, nullptr, LPH1, P(LP + 2), P(LP + 3), 1, 512, 1);
  gemv_k<<<256, 128, 0, stream>>>(P(LP + 4), LPH1, P(LP + 5), LPH2, 512);
  lnk(LPH2, nullptr, LPH2, P(LP + 6), P(LP + 7), 1, 256, 1);
  gemv_k<<<128, 128, 0, stream>>>(P(LP + 8), LPH2, P(LP + 9), LPH3, 256);
  lnk(LPH3, nullptr, LPH3, P(LP + 10), P(LP + 11), 1, 128, 1);
  lp_final<<<1, 128, 0, stream>>>(P(LP + 12), LPH3, P(LP + 13), (float*)d_out);
}
